// GTpytorch_84353157693914
// MI455X (gfx1250) — matrix operand order repaired and checked by reading, not compiled, not run on hardware
//
#include <hip/hip_runtime.h>
#include <math.h>
#include <stdint.h>

typedef __attribute__((ext_vector_type(16))) _Float16 v16h;
typedef __attribute__((ext_vector_type(8)))  _Float16 v8h;
typedef __attribute__((ext_vector_type(8)))  float    v8f;

#define DEPTH     6
#define N_EST     16
#define IN_DIM    512
#define N_CLASSES 100
#define BATCH     8192
#define N_NODES   63            // 2^6 - 1
#define N_LEAVES  64
#define KDIM      1024          // N_EST * N_LEAVES (GEMM K)
#define NPAD      112           // N_CLASSES padded to 7*16
#define NT        7             // NPAD / 16 output column tiles

// ---------------------------------------------------------------------------
// Kernel 1: per-node argmax over feature dim.
// entmax15 + hardmax == one-hot at argmax_d T[e,n,:] (monotone transform),
// so t_dot = max value of the row and the selected feature is its index.
// One wave32 per node (1008 nodes).
// ---------------------------------------------------------------------------
__global__ void node_argmax_kernel(const float* __restrict__ T,
                                   float* __restrict__ tval,
                                   int*   __restrict__ didx) {
  const int wave = (blockIdx.x * blockDim.x + threadIdx.x) >> 5;
  const int lane = threadIdx.x & 31;
  if (wave >= N_EST * N_NODES) return;
  const float* row = T + (size_t)wave * IN_DIM;
  float best = -3.4e38f;
  int   bidx = 0;
#pragma unroll
  for (int k = 0; k < IN_DIM / 32; ++k) {
    const int d = lane + 32 * k;
    const float v = row[d];
    if (v > best) { best = v; bidx = d; }
  }
#pragma unroll
  for (int off = 16; off > 0; off >>= 1) {
    const float ov = __shfl_down(best, off, 32);
    const int   oi = __shfl_down(bidx, off, 32);
    if (ov > best) { best = ov; bidx = oi; }
  }
  if (lane == 0) { tval[wave] = best; didx[wave] = bidx; }
}

// ---------------------------------------------------------------------------
// Kernel 2: build Lt[n][k] = f16(L[k/64][k%64][n]) for n<100, 0 for pad cols.
// N-major so a WMMA B-fragment lane reads 16 contiguous K values.
// ---------------------------------------------------------------------------
__global__ void pack_L_kernel(const float* __restrict__ L,
                              _Float16* __restrict__ Lt) {
  const int id = blockIdx.x * blockDim.x + threadIdx.x;   // n*KDIM + k
  if (id >= NPAD * KDIM) return;
  const int n = id / KDIM;
  const int k = id % KDIM;
  const float v = (n < N_CLASSES) ? L[(size_t)k * N_CLASSES + n] : 0.0f;
  Lt[id] = (_Float16)v;
}

// ---------------------------------------------------------------------------
// Kernel 3: per-(b,e) thread computes the 64 leaf "probabilities"
//   s[n] = floor(t[e,n] - x[b, d*[e,n]])
//   p[l] = prod_j (bit ? 1-s : s) over the 6 ancestors of leaf l
// Node tables cached in LDS; p stored f16 (exact: small integers).
// ---------------------------------------------------------------------------
__global__ void leaf_prob_kernel(const float* __restrict__ x,
                                 const float* __restrict__ tval,
                                 const int*   __restrict__ didx,
                                 _Float16* __restrict__ P) {
  __shared__ float s_t[N_EST * N_NODES];
  __shared__ int   s_d[N_EST * N_NODES];
  for (int i = threadIdx.x; i < N_EST * N_NODES; i += blockDim.x) {
    s_t[i] = tval[i];
    s_d[i] = didx[i];
  }
  __syncthreads();

  const int g = blockIdx.x * blockDim.x + threadIdx.x;  // b*N_EST + e
  const int b = g >> 4;
  const int e = g & 15;
  const float* xrow = x + (size_t)b * IN_DIM;

  float s[N_NODES];
#pragma unroll
  for (int n = 0; n < N_NODES; ++n) {
    const int node = e * N_NODES + n;
    s[n] = floorf(s_t[node] - xrow[s_d[node]]);
  }

  _Float16 ph[N_LEAVES];
#pragma unroll
  for (int l = 0; l < N_LEAVES; ++l) {
    float p = 1.0f;
#pragma unroll
    for (int j = 1; j <= DEPTH; ++j) {
      const int idx = (1 << (j - 1)) + (l >> (DEPTH - (j - 1))) - 1;
      const int bit = (l >> (DEPTH - j)) & 1;
      const float sv = s[idx];
      p *= bit ? (1.0f - sv) : sv;
    }
    ph[l] = (_Float16)p;
  }

  _Float16* dst = P + (size_t)g * N_LEAVES;   // == b*KDIM + e*64 (128B aligned)
#pragma unroll
  for (int i = 0; i < N_LEAVES / 8; ++i)
    reinterpret_cast<v8h*>(dst)[i] = reinterpret_cast<const v8h*>(ph)[i];
}

// ---------------------------------------------------------------------------
// Kernel 4: fused GEMM + softmax.
// One wave per 16-row batch tile computes the ENTIRE 16x112 output tile:
// 7 f32 accumulators (56 VGPRs), A fragment loaded once per K-step and
// amortized over 7 WMMAs (7 wmma : 16 b128-loads inner loop). Then the row
// softmax is done fully in-register: per C/D layout, lanes 0-15 hold rows
// v (VGPR v), lanes 16-31 hold rows v+8, columns live across lid=lane%16 and
// the 7 accumulator slots, so shfl_xor(8,4,2,1) reduces each row in-wave.
//
// Fragment layouts (wave32; C/D per ISA 7.12.2, A/B K order is this code's own):
//   A (16x32 f16): lane m=lane%16, half=lane/16 -> K in [half*16, +16)
//                  contiguous (two 16B loads)
//   B (32x16 f16): lane col n=lane%16, K in [half*16, +16) contiguous (32B)
//   C/D f32:       lane col n=lane%16, VGPR v -> row v + 8*half
// ---------------------------------------------------------------------------
__global__ void gemm_softmax_kernel(const _Float16* __restrict__ P,
                                    const _Float16* __restrict__ Lt,
                                    float* __restrict__ out) {
  const int wave = (blockIdx.x * blockDim.x + threadIdx.x) >> 5;  // mt
  const int lane = threadIdx.x & 31;
  if (wave >= BATCH / 16) return;
  const int mt   = wave;
  const int half = lane >> 4;                 // 0 or 1
  const int lid  = lane & 15;

  const _Float16* arow = P + (size_t)(mt * 16 + lid) * KDIM + half * 16;
  const _Float16* bcol[NT];
#pragma unroll
  for (int nt = 0; nt < NT; ++nt)
    bcol[nt] = Lt + (size_t)(nt * 16 + lid) * KDIM + half * 16;

  v8f c[NT];
#pragma unroll
  for (int nt = 0; nt < NT; ++nt) c[nt] = (v8f){};

#pragma unroll 2
  for (int kk = 0; kk < KDIM / 32; ++kk) {
    const int kb = kk * 32;
    const v8h alo = *reinterpret_cast<const v8h*>(arow + kb);
    const v8h ahi = *reinterpret_cast<const v8h*>(arow + kb + 8);
    v16h A;
#pragma unroll
    for (int i = 0; i < 8; ++i) { A[i] = alo[i]; A[i + 8] = ahi[i]; }
#pragma unroll
    for (int nt = 0; nt < NT; ++nt) {
      const v16h B = *reinterpret_cast<const v16h*>(bcol[nt] + kb);
      c[nt] = __builtin_amdgcn_wmma_f32_16x16x32_f16(
          /*neg_a=*/false, A, /*neg_b=*/false, B,
          /*c_mod=*/(short)0, c[nt], /*reuse_a=*/false, /*reuse_b=*/false);
    }
  }

  // ---- in-register row softmax over the 100 valid columns ----
#pragma unroll
  for (int v = 0; v < 8; ++v) {
    float row[NT];
#pragma unroll
    for (int nt = 0; nt < NT; ++nt) row[nt] = c[nt][v];

    // column nt*16+lid is valid iff nt<6, or nt==6 && lid<4 (cols 96..99)
    float m = -3.4e38f;
#pragma unroll
    for (int nt = 0; nt < NT; ++nt) {
      const bool valid = (nt < 6) | (lid < 4);
      if (valid) m = fmaxf(m, row[nt]);
    }
#pragma unroll
    for (int off = 8; off > 0; off >>= 1)
      m = fmaxf(m, __shfl_xor(m, off, 32));   // reduces within each 16-lane half

    float sum = 0.0f;
#pragma unroll
    for (int nt = 0; nt < NT; ++nt) {
      const bool valid = (nt < 6) | (lid < 4);
      if (valid) { row[nt] = expf(row[nt] - m); sum += row[nt]; }
    }
#pragma unroll
    for (int off = 8; off > 0; off >>= 1)
      sum += __shfl_xor(sum, off, 32);

    const float inv = 1.0f / sum;
    const int grow = mt * 16 + half * 8 + v;
    float* orow = out + (size_t)grow * N_CLASSES;
#pragma unroll
    for (int nt = 0; nt < NT; ++nt) {
      const int col = nt * 16 + lid;
      if (col < N_CLASSES) orow[col] = row[nt] * inv;
    }
  }
}

// ---------------------------------------------------------------------------
// Host-side launcher. Workspace layout (bytes):
//   [0,     4096)        tval  f32[1008]
//   [4096,  8192)        didx  i32[1008]
//   [8192,  256K)        Lt    f16[112*1024]   (229,376 B used)
//   [256K,  256K+16M)    P     f16[8192*1024]
// Total ~17 MB.
// ---------------------------------------------------------------------------
extern "C" void kernel_launch(void* const* d_in, const int* in_sizes, int n_in,
                              void* d_out, int out_size, void* d_ws, size_t ws_size,
                              hipStream_t stream) {
  (void)in_sizes; (void)n_in; (void)out_size; (void)ws_size;
  const float* x = (const float*)d_in[0];   // (8192, 512)
  const float* T = (const float*)d_in[1];   // (16, 63, 512)
  const float* L = (const float*)d_in[2];   // (16, 64, 100)

  char* ws = (char*)d_ws;
  float*    tval = (float*)(ws + 0);
  int*      didx = (int*)(ws + 4096);
  _Float16* Lt   = (_Float16*)(ws + 8192);
  _Float16* P    = (_Float16*)(ws + 262144);
  float*    out  = (float*)d_out;

  // 1) per-node argmax: 1008 waves, 8 waves/block
  node_argmax_kernel<<<(N_EST * N_NODES) / 8, 256, 0, stream>>>(T, tval, didx);
  // 2) pack L -> f16 transposed+padded
  pack_L_kernel<<<(NPAD * KDIM + 255) / 256, 256, 0, stream>>>(L, Lt);
  // 3) leaf probabilities: one thread per (b,e)
  leaf_prob_kernel<<<(BATCH * N_EST) / 256, 256, 0, stream>>>(x, tval, didx, P);
  // 4) fused GEMM + softmax: 512 row-tiles, 8 waves/block -> 64 blocks
  gemm_softmax_kernel<<<(BATCH / 16) / 8, 256, 0, stream>>>(P, Lt, out);
}
